// ContextualAttention_77893526880377
// MI455X (gfx1250) — matrix operand order repaired and checked by reading, not compiled, not run on hardware
//
#include <hip/hip_runtime.h>

typedef unsigned short u16;
typedef __attribute__((ext_vector_type(16))) __bf16 v16bf;
typedef __attribute__((ext_vector_type(8)))  float  v8f;
typedef __attribute__((ext_vector_type(4)))  int    v4i;

typedef __attribute__((address_space(1))) v4i* gas1_v4i;
typedef __attribute__((address_space(3))) v4i* las3_v4i;

union Frag {
    v16bf v;
    uint4 q[2];
};

#if __has_builtin(__builtin_amdgcn_global_load_async_to_lds_b128)
#define HAVE_ASYNC_LDS 1
#else
#define HAVE_ASYNC_LDS 0
#endif

#if HAVE_ASYNC_LDS
__device__ __forceinline__ void cp16_async(const u16* g, u16* l) {
    __builtin_amdgcn_global_load_async_to_lds_b128(
        (gas1_v4i)(v4i*)(g), (las3_v4i)(v4i*)(l), 0, 0);
}
#endif

__device__ __forceinline__ void wait_async_then_barrier() {
#if HAVE_ASYNC_LDS
#if __has_builtin(__builtin_amdgcn_s_wait_asynccnt)
    __builtin_amdgcn_s_wait_asynccnt(0);
#else
    asm volatile("s_wait_asynccnt 0x0" ::: "memory");
#endif
#endif
    __syncthreads();
}

__device__ __forceinline__ u16 f2bf(float f) {
    unsigned u = __float_as_uint(f);
    unsigned r = (u + 0x7FFFu + ((u >> 16) & 1u)) >> 16;
    return (u16)r;
}

__device__ __forceinline__ float rmax16(float v) {
    v = fmaxf(v, __shfl_xor(v, 1, 32));
    v = fmaxf(v, __shfl_xor(v, 2, 32));
    v = fmaxf(v, __shfl_xor(v, 4, 32));
    v = fmaxf(v, __shfl_xor(v, 8, 32));
    return v;
}
__device__ __forceinline__ float rsum16(float v) {
    v += __shfl_xor(v, 1, 32);
    v += __shfl_xor(v, 2, 32);
    v += __shfl_xor(v, 4, 32);
    v += __shfl_xor(v, 8, 32);
    return v;
}
__device__ __forceinline__ float wsum32(float v) {
    for (int m = 1; m < 32; m <<= 1) v += __shfl_xor(v, m, 32);
    return v;
}

// ---------------- bf16 conversion kernels ----------------

__global__ void cvt_kernel(const float* __restrict__ in, u16* __restrict__ out, int n) {
    int i = blockIdx.x * blockDim.x + threadIdx.x;
    if (i < n) out[i] = f2bf(in[i]);
}

// out: bf16 [B, T, E] with rows 0..TI-1 from x, TI..T-1 from context
// grid: (E/256, T, B)
__global__ void cvt_concat_kernel(const float* __restrict__ x, const float* __restrict__ ctx,
                                  u16* __restrict__ out, int TI_, int T_, int E_) {
    int e = blockIdx.x * blockDim.x + threadIdx.x;
    int t = blockIdx.y;
    int b = blockIdx.z;
    float v = (t < TI_) ? x[((size_t)b * TI_ + t) * E_ + e]
                        : ctx[((size_t)b * (T_ - TI_) + (t - TI_)) * E_ + e];
    out[((size_t)b * T_ + t) * E_ + e] = f2bf(v);
}

// V: f32 [B,T,H,D] -> bf16 [B,H,D,T]; grid: (T/256, D, B*H)
__global__ void vT_cvt_kernel(const float* __restrict__ in, u16* __restrict__ out,
                              int T_, int H_, int D_) {
    int t = blockIdx.x * blockDim.x + threadIdx.x;
    int d = blockIdx.y;
    int bh = blockIdx.z;
    int h = bh % H_;
    int b = bh / H_;
    float v = in[(((size_t)b * T_ + t) * H_ + h) * D_ + d];
    out[((size_t)bh * D_ + d) * T_ + t] = f2bf(v);
}

// ---------------- fused QK layernorm + scale + bf16, head-major output ----------------
// in : f32 [Rows, H, 64] ; out: bf16 [B, H, Tdim, 64] ; grid: (H, Tdim, B), block 32
__global__ __launch_bounds__(32)
void ln_cvt_kernel(const float* __restrict__ in, u16* __restrict__ out,
                   const float* __restrict__ gamma, const float* __restrict__ beta,
                   int Tdim, int H_, float rscale) {
    int h = blockIdx.x;
    int t = blockIdx.y;
    int b = blockIdx.z;
    int lane = threadIdx.x;
    const float* src = in + ((size_t)(b * Tdim + t) * H_ + h) * 64;
    float a = src[lane];
    float c = src[lane + 32];
    float mu = wsum32(a + c) * (1.0f / 64.0f);
    float da = a - mu, dc = c - mu;
    float var = wsum32(da * da + dc * dc) * (1.0f / 64.0f);
    float rstd = __frsqrt_rn(var + 1e-5f);
    u16* dst = out + (((size_t)(b * H_ + h) * Tdim) + t) * 64;
    dst[lane]      = f2bf((da * rstd * gamma[lane]      + beta[lane])      * rscale);
    dst[lane + 32] = f2bf((dc * rstd * gamma[lane + 32] + beta[lane + 32]) * rscale);
}

// ---------------- generic bf16 WMMA GEMM:  C[m,n] = sum_k A[m,k] * W[n,k] (+bias) ----------------
// Double-buffered LDS with async global->LDS staging (ASYNCcnt path) when available.
// A row m maps to global row (m/Mb)*Sb + m%Mb  (computed once per thread).
#define TILE_M 128
#define TILE_N 64
#define KC 32

__global__ __launch_bounds__(128)
void gemm_bf16_kernel(const u16* __restrict__ A, const u16* __restrict__ W,
                      float* __restrict__ C, const float* __restrict__ bias,
                      int N, int K, int Mb, int Sb) {
    __shared__ __align__(16) u16 lA[2][TILE_M * KC];
    __shared__ __align__(16) u16 lB[2][TILE_N * KC];
    int ntiles = N / TILE_N;
    int mBlk = blockIdx.x / ntiles;
    int nBlk = blockIdx.x % ntiles;
    int tid = threadIdx.x;
    int wave = tid >> 5;
    int lane = tid & 31;
    int hi = lane >> 4;
    int l16 = lane & 15;
    int mBase = mBlk * TILE_M;

    // staging assignments (hoisted): A tile = 128x32 bf16 = 512 16B-chunks -> 4/thread
    const u16* srcA[4];
    int dstA[4];
#pragma unroll
    for (int j = 0; j < 4; ++j) {
        int c = tid + j * 128;
        int row = c >> 2;           // 4 chunks per row
        int col = (c & 3) * 8;
        int gm = mBase + row;
        int b = gm / Mb, t = gm - b * Mb;
        srcA[j] = A + ((size_t)b * Sb + t) * K + col;
        dstA[j] = row * KC + col;
    }
    const u16* srcB[2];
    int dstB[2];
#pragma unroll
    for (int j = 0; j < 2; ++j) {
        int c = tid + j * 128;
        int row = c >> 2;
        int col = (c & 3) * 8;
        srcB[j] = W + (size_t)(nBlk * TILE_N + row) * K + col;
        dstB[j] = row * KC + col;
    }

    v8f acc[2][4] = {};

    // prologue: stage chunk 0 into buffer 0
#if HAVE_ASYNC_LDS
#pragma unroll
    for (int j = 0; j < 4; ++j) cp16_async(srcA[j], &lA[0][dstA[j]]);
#pragma unroll
    for (int j = 0; j < 2; ++j) cp16_async(srcB[j], &lB[0][dstB[j]]);
#else
    {
        uint4 ta[4], tb[2];
#pragma unroll
        for (int j = 0; j < 4; ++j) ta[j] = *(const uint4*)srcA[j];
#pragma unroll
        for (int j = 0; j < 2; ++j) tb[j] = *(const uint4*)srcB[j];
#pragma unroll
        for (int j = 0; j < 4; ++j) *(uint4*)&lA[0][dstA[j]] = ta[j];
#pragma unroll
        for (int j = 0; j < 2; ++j) *(uint4*)&lB[0][dstB[j]] = tb[j];
    }
#endif
    wait_async_then_barrier();

    int nIter = K / KC;
    for (int it = 0; it < nIter; ++it) {
        int cur = it & 1, nxt = cur ^ 1;
        if (it + 1 < nIter) {
            int ko = (it + 1) * KC;
#if HAVE_ASYNC_LDS
#pragma unroll
            for (int j = 0; j < 4; ++j) cp16_async(srcA[j] + ko, &lA[nxt][dstA[j]]);
#pragma unroll
            for (int j = 0; j < 2; ++j) cp16_async(srcB[j] + ko, &lB[nxt][dstB[j]]);
#else
            uint4 ta[4], tb[2];
#pragma unroll
            for (int j = 0; j < 4; ++j) ta[j] = *(const uint4*)(srcA[j] + ko);
#pragma unroll
            for (int j = 0; j < 2; ++j) tb[j] = *(const uint4*)(srcB[j] + ko);
#pragma unroll
            for (int j = 0; j < 4; ++j) *(uint4*)&lA[nxt][dstA[j]] = ta[j];
#pragma unroll
            for (int j = 0; j < 2; ++j) *(uint4*)&lB[nxt][dstB[j]] = tb[j];
#endif
        }
        // load all fragments, then issue all WMMAs
        Frag fa[2], fb[4];
#pragma unroll
        for (int s = 0; s < 2; ++s) {
            int row = wave * 32 + s * 16 + l16;
            int koff = hi ? 8 : 0;
            fa[s].q[0] = *(const uint4*)&lA[cur][row * KC + koff];
            fa[s].q[1] = *(const uint4*)&lA[cur][row * KC + koff + 16];
        }
#pragma unroll
        for (int t = 0; t < 4; ++t) {
            int col = t * 16 + l16;
            int koff = hi ? 8 : 0;
            fb[t].q[0] = *(const uint4*)&lB[cur][col * KC + koff];
            fb[t].q[1] = *(const uint4*)&lB[cur][col * KC + koff + 16];
        }
#pragma unroll
        for (int s = 0; s < 2; ++s)
#pragma unroll
            for (int t = 0; t < 4; ++t)
                acc[s][t] = __builtin_amdgcn_wmma_f32_16x16x32_bf16(
                    false, fa[s].v, false, fb[t].v, (short)0, acc[s][t], false, false);
        wait_async_then_barrier();
    }

    int gn0 = nBlk * TILE_N;
#pragma unroll
    for (int s = 0; s < 2; ++s)
#pragma unroll
        for (int t = 0; t < 4; ++t)
#pragma unroll
            for (int r = 0; r < 8; ++r) {
                int gm = mBase + wave * 32 + s * 16 + r + 8 * hi;
                int gc = gn0 + t * 16 + l16;
                float v = acc[s][t][r];
                if (bias) v += bias[gc];
                C[(size_t)gm * N + gc] = v;
            }
}

// ---------------- flash attention ----------------
// Q: bf16 [B,H,TI,64], K: bf16 [B,H,T,64], Vt: bf16 [B,H,64,T]
// Out: bf16 [B*TI, 1024]
__global__ __launch_bounds__(128)
void attn_kernel(const u16* __restrict__ Q, const u16* __restrict__ Km,
                 const u16* __restrict__ Vt, u16* __restrict__ Out,
                 int TI_, int T_, int H_) {
    __shared__ __align__(16) u16 Pl[4][16][64];
    int mtiles = TI_ / 64;
    int mBlk = blockIdx.x % mtiles;
    int bh = blockIdx.x / mtiles;
    int h = bh % H_;
    int b = bh / H_;
    int tid = threadIdx.x;
    int wave = tid >> 5;
    int lane = tid & 31;
    int hi = lane >> 4;
    int l16 = lane & 15;

    const u16* qbase = Q + (size_t)bh * TI_ * 64;
    const u16* kbase = Km + (size_t)bh * T_ * 64;
    const u16* vbase = Vt + (size_t)bh * 64 * T_;
    int mRow = mBlk * 64 + wave * 16;

    Frag fq[2];
#pragma unroll
    for (int kc2 = 0; kc2 < 2; ++kc2) {
        int row = mRow + l16;
        int koff = kc2 * 32 + (hi ? 8 : 0);
        fq[kc2].q[0] = *(const uint4*)(qbase + (size_t)row * 64 + koff);
        fq[kc2].q[1] = *(const uint4*)(qbase + (size_t)row * 64 + koff + 16);
    }

    v8f oacc[4] = {};
    float mrow[8], lrow[8];
#pragma unroll
    for (int r = 0; r < 8; ++r) { mrow[r] = -1e30f; lrow[r] = 0.0f; }

    for (int t0 = 0; t0 < T_; t0 += 64) {
        // prefetch next K/V chunk (global_prefetch_b8)
        if (t0 + 64 < T_) {
            __builtin_prefetch(kbase + (size_t)(t0 + 64 + lane * 2) * 64, 0, 1);
            __builtin_prefetch(vbase + (size_t)(lane * 2) * T_ + t0 + 64, 0, 1);
        }
        // S = Q K^T  (16 x 64 per wave)
        v8f sacc[4] = {};
#pragma unroll
        for (int kc2 = 0; kc2 < 2; ++kc2) {
            Frag fk[4];
#pragma unroll
            for (int t = 0; t < 4; ++t) {
                int krow = t0 + t * 16 + l16;
                int koff = kc2 * 32 + (hi ? 8 : 0);
                fk[t].q[0] = *(const uint4*)(kbase + (size_t)krow * 64 + koff);
                fk[t].q[1] = *(const uint4*)(kbase + (size_t)krow * 64 + koff + 16);
            }
#pragma unroll
            for (int t = 0; t < 4; ++t)
                sacc[t] = __builtin_amdgcn_wmma_f32_16x16x32_bf16(
                    false, fq[kc2].v, false, fk[t].v, (short)0, sacc[t], false, false);
        }
        // online softmax update, row-wise
#pragma unroll
        for (int r = 0; r < 8; ++r) {
            float cm = fmaxf(fmaxf(sacc[0][r], sacc[1][r]), fmaxf(sacc[2][r], sacc[3][r]));
            cm = rmax16(cm);
            float nm = fmaxf(mrow[r], cm);
            float alpha = __expf(mrow[r] - nm);
            mrow[r] = nm;
            float rs = 0.0f;
#pragma unroll
            for (int t = 0; t < 4; ++t) {
                float p = __expf(sacc[t][r] - nm);
                sacc[t][r] = p;
                rs += p;
            }
            rs = rsum16(rs);
            lrow[r] = lrow[r] * alpha + rs;
#pragma unroll
            for (int t = 0; t < 4; ++t) oacc[t][r] *= alpha;
        }
        // P (C-layout f32) -> LDS bf16, re-load in A-layout (wave-local, DS in-order)
#pragma unroll
        for (int t = 0; t < 4; ++t)
#pragma unroll
            for (int r = 0; r < 8; ++r)
                Pl[wave][r + 8 * hi][t * 16 + l16] = f2bf(sacc[t][r]);
        asm volatile("s_wait_dscnt 0x0" ::: "memory");
#pragma unroll
        for (int kc2 = 0; kc2 < 2; ++kc2) {
            Frag fp;
            int koff = kc2 * 32 + (hi ? 8 : 0);
            fp.q[0] = *(const uint4*)&Pl[wave][l16][koff];
            fp.q[1] = *(const uint4*)&Pl[wave][l16][koff + 16];
            Frag fv[4];
#pragma unroll
            for (int dt = 0; dt < 4; ++dt) {
                int drow = dt * 16 + l16;
                int tcol = t0 + kc2 * 32 + (hi ? 8 : 0);
                fv[dt].q[0] = *(const uint4*)(vbase + (size_t)drow * T_ + tcol);
                fv[dt].q[1] = *(const uint4*)(vbase + (size_t)drow * T_ + tcol + 16);
            }
#pragma unroll
            for (int dt = 0; dt < 4; ++dt)
                oacc[dt] = __builtin_amdgcn_wmma_f32_16x16x32_bf16(
                    false, fp.v, false, fv[dt].v, (short)0, oacc[dt], false, false);
        }
    }
#pragma unroll
    for (int r = 0; r < 8; ++r) {
        float inv = 1.0f / lrow[r];
        int gm = mRow + r + 8 * hi;
#pragma unroll
        for (int dt = 0; dt < 4; ++dt) {
            int gd = dt * 16 + l16;
            Out[(size_t)(b * TI_ + gm) * 1024 + h * 64 + gd] = f2bf(oacc[dt][r] * inv);
        }
    }
}

// ---------------- launch ----------------

extern "C" void kernel_launch(void* const* d_in, const int* in_sizes, int n_in,
                              void* d_out, int out_size, void* d_ws, size_t ws_size,
                              hipStream_t stream) {
    const float* x   = (const float*)d_in[0];
    const float* ctx = (const float*)d_in[1];
    const float* Wq  = (const float*)d_in[2];
    const float* Wk  = (const float*)d_in[3];
    const float* Wv  = (const float*)d_in[4];
    const float* Wu  = (const float*)d_in[5];
    const float* bu  = (const float*)d_in[6];
    const float* qg  = (const float*)d_in[7];
    const float* qb  = (const float*)d_in[8];
    const float* kg  = (const float*)d_in[9];
    const float* kb  = (const float*)d_in[10];
    float* out = (float*)d_out;

    const int B = 4, TI = 1024, T = 2048, E = 1024, H = 16, D = 64;
    const float RSCALE = 0.1767766953f;  // 1 / 1024^0.25

    char* ws = (char*)d_ws;
    size_t off = 0;
    u16* xc_bf = (u16*)(ws + off);  off += (size_t)B * T * E * 2;
    u16* Wq_bf = (u16*)(ws + off);  off += (size_t)E * E * 2;
    u16* Wk_bf = (u16*)(ws + off);  off += (size_t)E * E * 2;
    u16* Wv_bf = (u16*)(ws + off);  off += (size_t)E * E * 2;
    u16* Wu_bf = (u16*)(ws + off);  off += (size_t)E * E * 2;
    float* scratch = (float*)(ws + off); off += (size_t)B * T * E * 4;
    u16* q_bf   = (u16*)(ws + off); off += (size_t)B * TI * E * 2;
    u16* k_bf   = (u16*)(ws + off); off += (size_t)B * T * E * 2;
    u16* vT_bf  = (u16*)(ws + off); off += (size_t)B * T * E * 2;
    u16* att_bf = (u16*)(ws + off); off += (size_t)B * TI * E * 2;

    // 1) bf16 conversions
    cvt_concat_kernel<<<dim3(E / 256, T, B), 256, 0, stream>>>(x, ctx, xc_bf, TI, T, E);
    cvt_kernel<<<dim3(E * E / 256), 256, 0, stream>>>(Wq, Wq_bf, E * E);
    cvt_kernel<<<dim3(E * E / 256), 256, 0, stream>>>(Wk, Wk_bf, E * E);
    cvt_kernel<<<dim3(E * E / 256), 256, 0, stream>>>(Wv, Wv_bf, E * E);
    cvt_kernel<<<dim3(E * E / 256), 256, 0, stream>>>(Wu, Wu_bf, E * E);

    // 2) K projection + LN + head-major bf16
    gemm_bf16_kernel<<<dim3((B * T / TILE_M) * (E / TILE_N)), 128, 0, stream>>>(
        xc_bf, Wk_bf, scratch, nullptr, E, E, T, T);
    ln_cvt_kernel<<<dim3(H, T, B), 32, 0, stream>>>(scratch, k_bf, kg, kb, T, H, RSCALE);

    // 3) V projection + transpose to [B,H,D,T]
    gemm_bf16_kernel<<<dim3((B * T / TILE_M) * (E / TILE_N)), 128, 0, stream>>>(
        xc_bf, Wv_bf, scratch, nullptr, E, E, T, T);
    vT_cvt_kernel<<<dim3(T / 256, D, B * H), 256, 0, stream>>>(scratch, vT_bf, T, H, D);

    // 4) Q projection (rows pulled out of the concat buffer) + LN
    gemm_bf16_kernel<<<dim3((B * TI / TILE_M) * (E / TILE_N)), 128, 0, stream>>>(
        xc_bf, Wq_bf, scratch, nullptr, E, E, TI, T);
    ln_cvt_kernel<<<dim3(H, TI, B), 32, 0, stream>>>(scratch, q_bf, qg, qb, TI, H, RSCALE);

    // 5) flash attention
    attn_kernel<<<dim3(B * H * (TI / 64)), 128, 0, stream>>>(q_bf, k_bf, vT_bf, att_bf, TI, T, H);

    // 6) output projection with bias -> f32 d_out
    gemm_bf16_kernel<<<dim3((B * TI / TILE_M) * (E / TILE_N)), 128, 0, stream>>>(
        att_bf, Wu_bf, out, bu, E, E, TI, TI);
}
